// PIVRNN_89919435309694
// MI455X (gfx1250) — matrix operand order repaired and checked by reading, not compiled, not run on hardware
//
#include <hip/hip_runtime.h>
#include <hip/hip_bf16.h>
#include <math.h>

typedef _Float16 half_t;
typedef __attribute__((ext_vector_type(16))) _Float16 v16h;
typedef __attribute__((ext_vector_type(8)))  float    v8f;

#define NPL  11
#define NFE  6
#define CDIM 256
#define HDIM 512
#define VHD  256
#define ZD   64
#define TDD  2
#define BB   128
#define SS   512
#define K2B  128   // blocks for context kernel (per-block global scratch)

struct DevPtrs {
  // raw fp32 inputs
  const float *in, *tgt, *eps;
  const float *t1W1,*t1b1,*t1b2,*t2W1,*t2b1,*t2b2,*ctxb;
  const float *fwb0,*fwb1,*bwb0,*bwb1;
  const float *prb1,*prb2,*prmb,*prsb;
  const float *enb1,*enb2,*enmb,*ensb;
  const float *deb1,*deb2,*demW,*demb;
  // fp16 prepped weights, layout [out][Kpad], Kpad % 32 == 0
  const half_t *t1W2t,*t2W2t,*ctxWt;
  const half_t *fwWih0,*fwWhh0,*fwWih1,*fwWhh1;
  const half_t *bwWih0,*bwWhh0,*bwWih1,*bwWhh1;
  const half_t *prW1t,*prW2t,*prMS,*enW1t,*enW2t,*enMS,*deW1t,*deW2t;
  // activations / state (workspace)
  half_t *context,*hbw,*k2hbar,*k2st;
  half_t *h0,*h1,*hf0,*hf1;
  float  *c0,*c1,*cf0,*cf1,*Gg,*Gms,*pm,*ps_,*dm;
  half_t *zbuf,*actPR,*actEN,*actDE,*actFW,*actV1,*actV2;
  float  *outp;
};

union FragH { v16h v; unsigned int u[8]; uint4 q[2]; };

__device__ __forceinline__ float sigmf(float x){ return 1.0f/(1.0f+__expf(-x)); }

// D = A1*B1^T + A2*B2^T   (A: [M][lda] f16 row-major, B: [N][ldb] f16 rows = K-major)
// act==0: store raw f32 to Gout;  act==1: relu(.+bias)->f16 Hout;  act==2: (.+bias)->f16 Hout
// Fully inlined so all pointers keep global address space and all loop bounds
// are SGPR-resident -> scalar-branch loops, EXEC stays all-ones for every WMMA.
__device__ __forceinline__ void gemm2(
    const half_t* __restrict__ A1,int lda1,const half_t* __restrict__ B1,int ldb1,int K1,
    const half_t* __restrict__ A2,int lda2,const half_t* __restrict__ B2,int ldb2,int K2,
    int M,int N,
    float* __restrict__ Gout,int ldg,
    half_t* __restrict__ Hout,int ldh,const float* __restrict__ bias,int act)
{
  const int lane   = threadIdx.x & 31;
  // wave index is uniform per wave: force it into an SGPR so the tile loop is scalar
  const int wave   = __builtin_amdgcn_readfirstlane(threadIdx.x >> 5);
  const int nwaves = blockDim.x >> 5;
  const int laneN  = lane & 15;
  const int kh     = lane >> 4;
  const int ntn    = N >> 4;
  const int ntiles = (M >> 4) * ntn;
  for (int tile = wave; tile < ntiles; tile += nwaves) {
    const int mt = tile / ntn, nt = tile - mt*ntn;
    v8f acc = {0.f,0.f,0.f,0.f,0.f,0.f,0.f,0.f};
    for (int pass = 0; pass < 2; ++pass) {
      const half_t* A = pass ? A2 : A1;
      if (A == nullptr) continue;
      const half_t* Bw = pass ? B2 : B1;
      const int lda = pass ? lda2 : lda1;
      const int ldb = pass ? ldb2 : ldb1;
      const int K   = pass ? K2  : K1;
      const half_t* arow = A  + (size_t)(mt*16 + laneN)*lda + 8*kh;   // A frag: M=lane%16, K-half by lane/16
      const half_t* brow = Bw + (size_t)(nt*16 + laneN)*ldb + 8*kh;   // B frag: N=lane%16, K-half by lane/16
      __builtin_prefetch(brow, 0, 1);      // global_prefetch_b8 on the weight stream
      #pragma unroll 4
      for (int kb = 0; kb < K; kb += 32) {
        FragH a, b;
        a.q[0] = *reinterpret_cast<const uint4*>(arow + kb);        // K = 0..7  of this half
        a.q[1] = *reinterpret_cast<const uint4*>(arow + kb + 16);   // K = 16..23 of this half
        b.q[0] = *reinterpret_cast<const uint4*>(brow + kb);
        b.q[1] = *reinterpret_cast<const uint4*>(brow + kb + 16);
        acc = __builtin_amdgcn_wmma_f32_16x16x32_f16(false, a.v, false, b.v,
                                                     (short)0, acc, false, false);
      }
    }
    #pragma unroll
    for (int r = 0; r < 8; ++r) {
      const int m = mt*16 + r + 8*kh;
      const int n = nt*16 + laneN;
      float v = acc[r];
      if (act == 0) {
        Gout[(size_t)m*ldg + n] = v;
      } else {
        v += bias[n];
        if (act == 1 && v < 0.f) v = 0.f;
        Hout[(size_t)m*ldh + n] = (half_t)v;
      }
    }
  }
}

// i,f,g,o gate nonlinearity; G=[128][2048] f32, c f32 [128*512], h f16
__device__ __forceinline__ void lstm_gates(const float* __restrict__ G, const float* __restrict__ b,
                                           float* __restrict__ c, half_t* __restrict__ h,
                                           half_t* __restrict__ hcopy)
{
  for (int i = threadIdx.x; i < BB*HDIM; i += blockDim.x) {
    const int m = i >> 9, n = i & (HDIM-1);
    const float* g = G + (size_t)m*4*HDIM;
    float gi = g[n]          + b[n];
    float gf = g[HDIM+n]     + b[HDIM+n];
    float gg = g[2*HDIM+n]   + b[2*HDIM+n];
    float go = g[3*HDIM+n]   + b[3*HDIM+n];
    float cn = sigmf(gf)*c[i] + sigmf(gi)*tanhf(gg);
    c[i] = cn;
    half_t hn = (half_t)(sigmf(go)*tanhf(cn));
    h[i] = hn;
    if (hcopy) hcopy[i] = hn;
  }
}

// ---------------- weight prep ----------------
__global__ void k_copy_pad(half_t* __restrict__ dst, const float* __restrict__ src,
                           int OUT, int K, int Kpad) {
  int i = blockIdx.x*blockDim.x + threadIdx.x;
  if (i >= OUT*Kpad) return;
  int n = i / Kpad, k = i - n*Kpad;
  dst[i] = (k < K) ? (half_t)src[(size_t)n*K + k] : (half_t)0.f;
}
__global__ void k_trans_pad(half_t* __restrict__ dst, const float* __restrict__ src,
                            int OUT, int K, int Kpad) {
  int i = blockIdx.x*blockDim.x + threadIdx.x;
  if (i >= OUT*Kpad) return;
  int n = i / Kpad, k = i - n*Kpad;
  dst[i] = (k < K) ? (half_t)src[(size_t)k*OUT + n] : (half_t)0.f;
}

// ---------------- phase 1: team MLPs + context (parallel over s) ----------------
__global__ __launch_bounds__(256) void k_ctx(DevPtrs P)
{
  __shared__ float xsl[BB*(NPL*NFE)];
  const int tid = threadIdx.x;
  half_t* hbar = P.k2hbar + (size_t)blockIdx.x*BB*CDIM;
  half_t* stb  = P.k2st   + (size_t)blockIdx.x*BB*(2*CDIM);
  for (int s = blockIdx.x; s < SS; s += gridDim.x) {
    for (int team = 0; team < 2; ++team) {
      const float*  W1  = team ? P.t2W1  : P.t1W1;
      const float*  b1  = team ? P.t2b1  : P.t1b1;
      const float*  b2f = team ? P.t2b2  : P.t1b2;
      const half_t* W2t = team ? P.t2W2t : P.t1W2t;
      const int foff = team ? NPL*NFE : 0;
      for (int i = tid; i < BB*NPL*NFE; i += blockDim.x) {
        const int m = i / (NPL*NFE), f = i - m*(NPL*NFE);
        xsl[i] = P.in[(size_t)m*SS*(2*NPL*NFE) + (size_t)s*(2*NPL*NFE) + foff + f];
      }
      __syncthreads();
      for (int i = tid; i < BB*CDIM; i += blockDim.x) {
        const int m = i >> 8, cch = i & 255;
        float accm = 0.f;
        for (int p = 0; p < NPL; ++p) {
          float a = b1[cch];
          #pragma unroll
          for (int k = 0; k < NFE; ++k)
            a += xsl[m*(NPL*NFE) + p*NFE + k] * W1[k*CDIM + cch];
          accm += fmaxf(a, 0.f);
        }
        hbar[i] = (half_t)(accm * (1.0f/NPL));
      }
      __syncthreads();
      gemm2(hbar, CDIM, W2t, CDIM, CDIM,  nullptr,0,nullptr,0,0,
            BB, CDIM,  nullptr,0,  stb + team*CDIM, 2*CDIM, b2f, 1);
      __syncthreads();
    }
    gemm2(stb, 2*CDIM, P.ctxWt, 2*CDIM, 2*CDIM,  nullptr,0,nullptr,0,0,
          BB, CDIM,  nullptr,0,  P.context + (size_t)s*BB*CDIM, CDIM, P.ctxb, 1);
    __syncthreads();
  }
}

// ---------------- phase 2: backward LSTM (persistent, sequential t = S-1..0) ----------------
__global__ __launch_bounds__(1024) void k_bw(DevPtrs P)
{
  for (int i = threadIdx.x; i < BB*HDIM; i += blockDim.x) {
    P.h0[i] = (half_t)0.f; P.h1[i] = (half_t)0.f; P.c0[i] = 0.f; P.c1[i] = 0.f;
  }
  __syncthreads();
  for (int t = SS-1; t >= 0; --t) {
    const half_t* ct = P.context + (size_t)t*BB*CDIM;
    gemm2(ct, CDIM, P.bwWih0, CDIM, CDIM,  P.h0, HDIM, P.bwWhh0, HDIM, HDIM,
          BB, 4*HDIM,  P.Gg, 4*HDIM,  nullptr,0,nullptr, 0);
    __syncthreads();
    lstm_gates(P.Gg, P.bwb0, P.c0, P.h0, nullptr);
    __syncthreads();
    gemm2(P.h0, HDIM, P.bwWih1, HDIM, HDIM,  P.h1, HDIM, P.bwWhh1, HDIM, HDIM,
          BB, 4*HDIM,  P.Gg, 4*HDIM,  nullptr,0,nullptr, 0);
    __syncthreads();
    lstm_gates(P.Gg, P.bwb1, P.c1, P.h1, P.hbw + (size_t)t*BB*HDIM);
    __syncthreads();
  }
}

// ---------------- phase 3: forward VRNN loop (persistent, sequential t = 0..S-1) ----------------
__global__ __launch_bounds__(1024) void k_fw(DevPtrs P)
{
  __shared__ double red[1024];
  const int tid = threadIdx.x;
  for (int i = tid; i < BB*HDIM; i += blockDim.x) {
    P.hf0[i] = (half_t)0.f; P.hf1[i] = (half_t)0.f; P.cf0[i] = 0.f; P.cf1[i] = 0.f;
  }
  __syncthreads();
  double kacc = 0.0, nacc = 0.0, dacc = 0.0;
  const float ps0 = 108.f, ps1 = 72.f;
  const float nll_c = -3.6862316527834187f;  // ln(0.01)+0.5*ln(2*pi)
  for (int t = 0; t < SS; ++t) {
    const half_t* ct = P.context + (size_t)t*BB*CDIM;
    const half_t* hb = P.hbw     + (size_t)t*BB*HDIM;
    // actPR = [ct | hf1 | hb]  (128 x 1280)
    for (int i = tid; i < BB*1280; i += blockDim.x) {
      int m = i / 1280, c = i - m*1280;
      half_t v;
      if      (c < 256) v = ct[m*CDIM + c];
      else if (c < 768) v = P.hf1[m*HDIM + c - 256];
      else              v = hb[m*HDIM + c - 768];
      P.actPR[i] = v;
    }
    // actEN = [ct | tt | hf1 | hb | pad]  (128 x 1312)
    for (int i = tid; i < BB*1312; i += blockDim.x) {
      int m = i / 1312, c = i - m*1312;
      half_t v;
      if      (c < 256)  v = ct[m*CDIM + c];
      else if (c < 258)  v = (half_t)(P.tgt[(size_t)m*SS*TDD + t*TDD + (c-256)]
                                      * (c == 256 ? (1.f/108.f) : (1.f/72.f)));
      else if (c < 770)  v = P.hf1[m*HDIM + c - 258];
      else if (c < 1282) v = hb[m*HDIM + c - 770];
      else               v = (half_t)0.f;
      P.actEN[i] = v;
    }
    __syncthreads();
    // prior: mlp2 -> [mW|sW]
    gemm2(P.actPR,1280, P.prW1t,1280,1280, nullptr,0,nullptr,0,0, BB, VHD, nullptr,0, P.actV1, VHD, P.prb1, 1);
    __syncthreads();
    gemm2(P.actV1,VHD,  P.prW2t,VHD,VHD,   nullptr,0,nullptr,0,0, BB, VHD, nullptr,0, P.actV2, VHD, P.prb2, 1);
    __syncthreads();
    gemm2(P.actV2,VHD,  P.prMS, VHD,VHD,   nullptr,0,nullptr,0,0, BB, 2*ZD, P.Gms, 2*ZD, nullptr,0,nullptr, 0);
    __syncthreads();
    for (int i = tid; i < BB*ZD; i += blockDim.x) {
      int m = i >> 6, k = i & 63;
      P.pm[i] = P.Gms[m*2*ZD + k] + P.prmb[k];
      float x = P.Gms[m*2*ZD + ZD + k] + P.prsb[k];
      P.ps_[i] = (x > 20.f) ? x : log1pf(__expf(x));
    }
    __syncthreads();
    // encoder + z + kld
    gemm2(P.actEN,1312, P.enW1t,1312,1312, nullptr,0,nullptr,0,0, BB, VHD, nullptr,0, P.actV1, VHD, P.enb1, 1);
    __syncthreads();
    gemm2(P.actV1,VHD,  P.enW2t,VHD,VHD,   nullptr,0,nullptr,0,0, BB, VHD, nullptr,0, P.actV2, VHD, P.enb2, 1);
    __syncthreads();
    gemm2(P.actV2,VHD,  P.enMS, VHD,VHD,   nullptr,0,nullptr,0,0, BB, 2*ZD, P.Gms, 2*ZD, nullptr,0,nullptr, 0);
    __syncthreads();
    for (int i = tid; i < BB*ZD; i += blockDim.x) {
      int m = i >> 6, k = i & 63;
      float em = P.Gms[m*2*ZD + k] + P.enmb[k];
      float x  = P.Gms[m*2*ZD + ZD + k] + P.ensb[k];
      float es = (x > 20.f) ? x : log1pf(__expf(x));
      float z  = em + es * P.eps[(size_t)t*BB*ZD + i];
      P.zbuf[i] = (half_t)z;
      float pmv = P.pm[i], pstd = P.ps_[i];
      float dmu = em - pmv;
      kacc += (double)(logf(pstd) - logf(es) + (es*es + dmu*dmu)/(2.f*pstd*pstd) - 0.5f);
    }
    __syncthreads();
    // actDE = [ct | z | hf1 | hb]  (128 x 1344)
    for (int i = tid; i < BB*1344; i += blockDim.x) {
      int m = i / 1344, c = i - m*1344;
      half_t v;
      if      (c < 256) v = ct[m*CDIM + c];
      else if (c < 320) v = P.zbuf[m*ZD + c - 256];
      else if (c < 832) v = P.hf1[m*HDIM + c - 320];
      else              v = hb[m*HDIM + c - 832];
      P.actDE[i] = v;
    }
    // actFW = [ct | tt | z | pad]  (128 x 352)
    for (int i = tid; i < BB*352; i += blockDim.x) {
      int m = i / 352, c = i - m*352;
      half_t v;
      if      (c < 256) v = ct[m*CDIM + c];
      else if (c < 258) v = (half_t)(P.tgt[(size_t)m*SS*TDD + t*TDD + (c-256)]
                                     * (c == 256 ? (1.f/108.f) : (1.f/72.f)));
      else if (c < 322) v = P.zbuf[m*ZD + c - 258];
      else              v = (half_t)0.f;
      P.actFW[i] = v;
    }
    __syncthreads();
    // decoder
    gemm2(P.actDE,1344, P.deW1t,1344,1344, nullptr,0,nullptr,0,0, BB, VHD, nullptr,0, P.actV1, VHD, P.deb1, 1);
    __syncthreads();
    gemm2(P.actV1,VHD,  P.deW2t,VHD,VHD,   nullptr,0,nullptr,0,0, BB, VHD, nullptr,0, P.actV2, VHD, P.deb2, 1);
    __syncthreads();
    for (int i = tid; i < BB*TDD; i += blockDim.x) {   // dm: tiny N=2, VALU dot
      int m = i >> 1, d = i & 1;
      float s = P.demb[d];
      for (int k = 0; k < VHD; ++k)
        s += (float)P.actV2[m*VHD + k] * P.demW[k*TDD + d];
      P.dm[i] = s;
    }
    __syncthreads();
    for (int m = tid; m < BB; m += blockDim.x) {
      float t0 = P.tgt[(size_t)m*SS*TDD + t*TDD + 0] * (1.f/108.f);
      float t1 = P.tgt[(size_t)m*SS*TDD + t*TDD + 1] * (1.f/72.f);
      float d0 = t0 - P.dm[m*2+0];
      float d1 = t1 - P.dm[m*2+1];
      nacc += (double)((d0*d0 + d1*d1)*5000.f + 2.f*nll_c);
      float e0 = d0*ps0, e1 = d1*ps1;
      dacc += (double)sqrtf(e0*e0 + e1*e1);
    }
    // forward LSTM step
    gemm2(P.actFW,352, P.fwWih0,352,352,  P.hf0,HDIM, P.fwWhh0,HDIM,HDIM,
          BB, 4*HDIM,  P.Gg, 4*HDIM,  nullptr,0,nullptr, 0);
    __syncthreads();
    lstm_gates(P.Gg, P.fwb0, P.cf0, P.hf0, nullptr);
    __syncthreads();
    gemm2(P.hf0,HDIM, P.fwWih1,HDIM,HDIM,  P.hf1,HDIM, P.fwWhh1,HDIM,HDIM,
          BB, 4*HDIM,  P.Gg, 4*HDIM,  nullptr,0,nullptr, 0);
    __syncthreads();
    lstm_gates(P.Gg, P.fwb1, P.cf1, P.hf1, nullptr);
    __syncthreads();
  }
  // block reduction of the three scalars
  red[tid] = kacc; __syncthreads();
  for (int o = 512; o > 0; o >>= 1) { if (tid < o) red[tid] += red[tid+o]; __syncthreads(); }
  double ktot = red[0]; __syncthreads();
  red[tid] = nacc; __syncthreads();
  for (int o = 512; o > 0; o >>= 1) { if (tid < o) red[tid] += red[tid+o]; __syncthreads(); }
  double ntot = red[0]; __syncthreads();
  red[tid] = dacc; __syncthreads();
  for (int o = 512; o > 0; o >>= 1) { if (tid < o) red[tid] += red[tid+o]; __syncthreads(); }
  if (tid == 0) {
    P.outp[0] = (float)(ktot / (double)BB);
    P.outp[1] = (float)(ntot / (double)BB);
    P.outp[2] = (float)(red[0] / (double)(BB*SS));
  }
}

extern "C" void kernel_launch(void* const* d_in, const int* in_sizes, int n_in,
                              void* d_out, int out_size, void* d_ws, size_t ws_size,
                              hipStream_t stream)
{
  (void)in_sizes; (void)n_in; (void)out_size; (void)ws_size;
  const float* const* F = (const float* const*)d_in;

  DevPtrs P;
  P.in = F[0]; P.tgt = F[1]; P.eps = F[2];
  P.t1W1 = F[3];  P.t1b1 = F[4];  P.t1b2 = F[6];
  P.t2W1 = F[7];  P.t2b1 = F[8];  P.t2b2 = F[10];
  P.ctxb = F[12];
  P.fwb0 = F[15]; P.fwb1 = F[18]; P.bwb0 = F[21]; P.bwb1 = F[24];
  P.prb1 = F[26]; P.prb2 = F[28]; P.prmb = F[30]; P.prsb = F[32];
  P.enb1 = F[34]; P.enb2 = F[36]; P.enmb = F[38]; P.ensb = F[40];
  P.deb1 = F[42]; P.deb2 = F[44]; P.demW = F[45]; P.demb = F[46];

  char* w = (char*)d_ws;
  auto allocH = [&](size_t n)->half_t* { half_t* p=(half_t*)w; w += ((n*2+255)&~(size_t)255); return p; };
  auto allocF = [&](size_t n)->float*  { float*  p=(float*)w;  w += ((n*4+255)&~(size_t)255); return p; };

  half_t* t1W2t  = allocH(256*256);
  half_t* t2W2t  = allocH(256*256);
  half_t* ctxWt  = allocH(256*512);
  half_t* fwWih0 = allocH((size_t)2048*352);
  half_t* fwWhh0 = allocH((size_t)2048*512);
  half_t* fwWih1 = allocH((size_t)2048*512);
  half_t* fwWhh1 = allocH((size_t)2048*512);
  half_t* bwWih0 = allocH((size_t)2048*256);
  half_t* bwWhh0 = allocH((size_t)2048*512);
  half_t* bwWih1 = allocH((size_t)2048*512);
  half_t* bwWhh1 = allocH((size_t)2048*512);
  half_t* prW1t  = allocH((size_t)256*1280);
  half_t* prW2t  = allocH(256*256);
  half_t* prMS   = allocH(128*256);
  half_t* enW1t  = allocH((size_t)256*1312);
  half_t* enW2t  = allocH(256*256);
  half_t* enMS   = allocH(128*256);
  half_t* deW1t  = allocH((size_t)256*1344);
  half_t* deW2t  = allocH(256*256);

  P.t1W2t=t1W2t; P.t2W2t=t2W2t; P.ctxWt=ctxWt;
  P.fwWih0=fwWih0; P.fwWhh0=fwWhh0; P.fwWih1=fwWih1; P.fwWhh1=fwWhh1;
  P.bwWih0=bwWih0; P.bwWhh0=bwWhh0; P.bwWih1=bwWih1; P.bwWhh1=bwWhh1;
  P.prW1t=prW1t; P.prW2t=prW2t; P.prMS=prMS;
  P.enW1t=enW1t; P.enW2t=enW2t; P.enMS=enMS;
  P.deW1t=deW1t; P.deW2t=deW2t;

  P.context = allocH((size_t)SS*BB*CDIM);
  P.hbw     = allocH((size_t)SS*BB*HDIM);
  P.k2hbar  = allocH((size_t)K2B*BB*CDIM);
  P.k2st    = allocH((size_t)K2B*BB*2*CDIM);
  P.h0  = allocH(BB*HDIM); P.h1  = allocH(BB*HDIM);
  P.hf0 = allocH(BB*HDIM); P.hf1 = allocH(BB*HDIM);
  P.c0  = allocF(BB*HDIM); P.c1  = allocF(BB*HDIM);
  P.cf0 = allocF(BB*HDIM); P.cf1 = allocF(BB*HDIM);
  P.Gg  = allocF((size_t)BB*4*HDIM);
  P.Gms = allocF(BB*2*ZD);
  P.pm  = allocF(BB*ZD);  P.ps_ = allocF(BB*ZD);
  P.dm  = allocF(BB*TDD);
  P.zbuf  = allocH(BB*ZD);
  P.actPR = allocH((size_t)BB*1280);
  P.actEN = allocH((size_t)BB*1312);
  P.actDE = allocH((size_t)BB*1344);
  P.actFW = allocH((size_t)BB*352);
  P.actV1 = allocH(BB*VHD);
  P.actV2 = allocH(BB*VHD);
  P.outp  = (float*)d_out;

  auto copyW = [&](half_t* dst, const float* src, int OUT, int K, int Kpad){
    int tot = OUT*Kpad;
    k_copy_pad<<<(tot+255)/256, 256, 0, stream>>>(dst, src, OUT, K, Kpad);
  };
  auto transW = [&](half_t* dst, const float* src, int OUT, int K, int Kpad){
    int tot = OUT*Kpad;
    k_trans_pad<<<(tot+255)/256, 256, 0, stream>>>(dst, src, OUT, K, Kpad);
  };

  // [in][out] weights -> transpose to [out][Kpad]
  transW(t1W2t, F[5],  256, 256, 256);
  transW(t2W2t, F[9],  256, 256, 256);
  transW(ctxWt, F[11], 256, 512, 512);
  transW(prW1t, F[25], 256, 1280, 1280);
  transW(prW2t, F[27], 256, 256, 256);
  transW(prMS,            F[29], 64, 256, 256);
  transW(prMS + 64*256,   F[31], 64, 256, 256);
  transW(enW1t, F[33], 256, 1282, 1312);
  transW(enW2t, F[35], 256, 256, 256);
  transW(enMS,            F[37], 64, 256, 256);
  transW(enMS + 64*256,   F[39], 64, 256, 256);
  transW(deW1t, F[41], 256, 1344, 1344);
  transW(deW2t, F[43], 256, 256, 256);
  // [out][in] weights -> copy with K padding
  copyW(fwWih0, F[13], 2048, 322, 352);
  copyW(fwWhh0, F[14], 2048, 512, 512);
  copyW(fwWih1, F[16], 2048, 512, 512);
  copyW(fwWhh1, F[17], 2048, 512, 512);
  copyW(bwWih0, F[19], 2048, 256, 256);
  copyW(bwWhh0, F[20], 2048, 512, 512);
  copyW(bwWih1, F[22], 2048, 512, 512);
  copyW(bwWhh1, F[23], 2048, 512, 512);

  k_ctx<<<K2B, 256, 0, stream>>>(P);
  k_bw <<<1, 1024, 0, stream>>>(P);
  k_fw <<<1, 1024, 0, stream>>>(P);
}
